// SkimMemory_66279935311941
// MI455X (gfx1250) — matrix operand order repaired and checked by reading, not compiled, not run on hardware
//
#include <hip/hip_runtime.h>

// ---------- types ----------
typedef __attribute__((ext_vector_type(16))) __bf16        v16bf;
typedef __attribute__((ext_vector_type(8)))  float         v8f;
typedef __attribute__((ext_vector_type(8)))  unsigned int  v8u;   // 32B
typedef __attribute__((ext_vector_type(4)))  unsigned int  u32x4; // 16B

__device__ __forceinline__ unsigned short f2bf(float f) {
    unsigned int x = __builtin_bit_cast(unsigned int, f);
    unsigned int r = x + 0x7fffu + ((x >> 16) & 1u);   // RNE
    return (unsigned short)(r >> 16);
}
__device__ __forceinline__ float bf2f(unsigned short h) {
    return __builtin_bit_cast(float, ((unsigned int)h) << 16);
}
__device__ __forceinline__ v16bf make_a_frag(const u32x4 lo, const u32x4 hi) {
    v8u r;
    r[0]=lo[0]; r[1]=lo[1]; r[2]=lo[2]; r[3]=lo[3];
    r[4]=hi[0]; r[5]=hi[1]; r[6]=hi[2]; r[7]=hi[3];
    return __builtin_bit_cast(v16bf, r);
}

// ---------- fp32 -> bf16 convert ----------
__global__ void __launch_bounds__(256)
cvt_f32_bf16(const float* __restrict__ src, unsigned short* __restrict__ dst, int n) {
    for (int i = blockIdx.x * 256 + threadIdx.x; i < n; i += gridDim.x * 256)
        dst[i] = f2bf(src[i]);
}

// ---------- generic GEMM:  C[M,N] = scale * (A[M,K] @ W[N,K]^T), bf16 WMMA ----------
// Block = 256 threads (8 waves), tile = 64(M) x 128(N); wave = 16(M) x 64(N) = 4 wmma tiles.
template <bool OUT_BF16>
__global__ void __launch_bounds__(256)
gemm_xwt(const unsigned short* __restrict__ A, const unsigned short* __restrict__ W,
         void* __restrict__ Cout, int M, int N, int K, float scale) {
    const int lane = threadIdx.x & 31;
    const int wave = threadIdx.x >> 5;
    const int wm = wave & 3, wn = wave >> 2;
    const int h = lane >> 4, l16 = lane & 15;
    const int bm = blockIdx.y * 64 + wm * 16;
    const int bn = blockIdx.x * 128 + wn * 64;

    v8f c0 = {}, c1 = {}, c2 = {}, c3 = {};
    const unsigned short* arow = A + (size_t)(bm + l16) * K;
    const unsigned short* wb   = W + (size_t)bn * K;

    for (int ks = 0; ks < K; ks += 32) {
        u32x4 alo = *(const u32x4*)(arow + ks + h * 16);
        u32x4 ahi = *(const u32x4*)(arow + ks + h * 16 + 8);
        v16bf a = make_a_frag(alo, ahi);

        v16bf b0 = __builtin_bit_cast(v16bf, *(const v8u*)(wb + (size_t)( 0 + l16) * K + ks + h * 16));
        v16bf b1 = __builtin_bit_cast(v16bf, *(const v8u*)(wb + (size_t)(16 + l16) * K + ks + h * 16));
        v16bf b2 = __builtin_bit_cast(v16bf, *(const v8u*)(wb + (size_t)(32 + l16) * K + ks + h * 16));
        v16bf b3 = __builtin_bit_cast(v16bf, *(const v8u*)(wb + (size_t)(48 + l16) * K + ks + h * 16));

        c0 = __builtin_amdgcn_wmma_f32_16x16x32_bf16(false, a, false, b0, (short)0, c0, false, false);
        c1 = __builtin_amdgcn_wmma_f32_16x16x32_bf16(false, a, false, b1, (short)0, c1, false, false);
        c2 = __builtin_amdgcn_wmma_f32_16x16x32_bf16(false, a, false, b2, (short)0, c2, false, false);
        c3 = __builtin_amdgcn_wmma_f32_16x16x32_bf16(false, a, false, b3, (short)0, c3, false, false);
    }

#pragma unroll
    for (int r = 0; r < 8; ++r) {
        const int m = bm + r + 8 * h;
        const size_t base = (size_t)m * N;
#define STORE_C(creg, nt)                                                     \
        do {                                                                  \
            const int n = bn + (nt) * 16 + l16;                               \
            float vv = (creg)[r] * scale;                                     \
            if (OUT_BF16) ((unsigned short*)Cout)[base + n] = f2bf(vv);       \
            else          ((float*)Cout)[base + n] = vv;                      \
        } while (0)
        STORE_C(c0, 0); STORE_C(c1, 1); STORE_C(c2, 2); STORE_C(c3, 3);
#undef STORE_C
    }
}

// ---------- fused: scores (WMMA) + streaming top-32 + softmax + v-gather ----------
// grid.x = 2048/16 = 128 row-tiles; block = 256 threads (8 waves).
// Each chunk: 16 rows x 128 cols of scores; each wave owns one 16x16 tile.
__global__ void __launch_bounds__(256)
skim_attn(const unsigned short* __restrict__ q,   // [2048,1024] bf16, pre-scaled by 1/32
          const unsigned short* __restrict__ k,   // [32768,1024] bf16
          const unsigned short* __restrict__ v,   // [32768,1024] bf16
          unsigned short* __restrict__ ctx) {     // [2048,1024] bf16
    __shared__ unsigned short s_q[16 * 1024];     // 32 KB q tile
    __shared__ float s_sc[16 * 128];              // 8 KB score chunk
    __shared__ float s_topv[16 * 32];
    __shared__ int   s_topi[16 * 32];

    const int tid = threadIdx.x;
    const int lane = tid & 31, wave = tid >> 5;
    const int h = lane >> 4, l16 = lane & 15;
    const int rowbase = blockIdx.x * 16;

    { // stage q tile into LDS (as dwords)
        const unsigned int* qg = (const unsigned int*)(q + (size_t)rowbase * 1024);
        unsigned int* sq = (unsigned int*)s_q;
        for (int i = tid; i < 16 * 512; i += 256) sq[i] = qg[i];
    }
    for (int i = tid; i < 16 * 32; i += 256) { s_topv[i] = -3.0e38f; s_topi[i] = 0; }
    __syncthreads();

    for (int nb = 0; nb < 32768; nb += 128) {
        // ---- score tile via WMMA (A from LDS, B = k rows from L2) ----
        v8f c = {};
        const unsigned short* krow = k + (size_t)(nb + wave * 16 + l16) * 1024;
        const unsigned short* qrow = s_q + l16 * 1024;
        for (int ks = 0; ks < 1024; ks += 32) {
            u32x4 alo = *(const u32x4*)(qrow + ks + h * 16);
            u32x4 ahi = *(const u32x4*)(qrow + ks + h * 16 + 8);
            v16bf a = make_a_frag(alo, ahi);
            v16bf b = __builtin_bit_cast(v16bf, *(const v8u*)(krow + ks + h * 16));
            c = __builtin_amdgcn_wmma_f32_16x16x32_bf16(false, a, false, b, (short)0, c, false, false);
        }
#pragma unroll
        for (int r = 0; r < 8; ++r)
            s_sc[(r + 8 * h) * 128 + wave * 16 + l16] = c[r];
        __syncthreads();

        // ---- streaming top-32 update: 2 rows per wave, slot = lane ----
        for (int r2 = 0; r2 < 2; ++r2) {
            const int m = wave * 2 + r2;
            float tv = s_topv[m * 32 + lane];
            int   ti = s_topi[m * 32 + lane];
#pragma unroll
            for (int p = 0; p < 4; ++p) {
                const int j = p * 32 + lane;
                const float val = s_sc[m * 128 + j];
                const int gcol = nb + j;
                // current min slot (threshold)
                float mv = tv; int ms = lane;
#pragma unroll
                for (int off = 16; off; off >>= 1) {
                    float ov = __shfl_xor(mv, off);
                    int   os = __shfl_xor(ms, off);
                    if (ov < mv || (ov == mv && os < ms)) { mv = ov; ms = os; }
                }
                unsigned long long mask = __ballot(val > mv);
                while (mask) {
                    const int src = __ffsll((long long)mask) - 1;
                    mask &= mask - 1;
                    const float cv = __shfl(val, src);
                    const int   ci = __shfl(gcol, src);
                    mv = tv; ms = lane;
#pragma unroll
                    for (int off = 16; off; off >>= 1) {
                        float ov = __shfl_xor(mv, off);
                        int   os = __shfl_xor(ms, off);
                        if (ov < mv || (ov == mv && os < ms)) { mv = ov; ms = os; }
                    }
                    if (cv > mv && lane == ms) { tv = cv; ti = ci; }
                }
            }
            s_topv[m * 32 + lane] = tv;
            s_topi[m * 32 + lane] = ti;
        }
        __syncthreads();
    }

    // ---- softmax over 32 kept scores (2 rows per wave) ----
    for (int r2 = 0; r2 < 2; ++r2) {
        const int m = wave * 2 + r2;
        float x = s_topv[m * 32 + lane];
        float mx = x;
#pragma unroll
        for (int off = 16; off; off >>= 1) mx = fmaxf(mx, __shfl_xor(mx, off));
        float e = __expf(x - mx);
        float s = e;
#pragma unroll
        for (int off = 16; off; off >>= 1) s += __shfl_xor(s, off);
        s_topv[m * 32 + lane] = e / s;
    }
    __syncthreads();

    // ---- ctx[m,:] = sum_t attn[t] * v[idx[t],:]  (lane owns 32 contiguous cols) ----
    for (int r2 = 0; r2 < 2; ++r2) {
        const int m = wave * 2 + r2;
        float acc[32];
#pragma unroll
        for (int c2 = 0; c2 < 32; ++c2) acc[c2] = 0.0f;
        for (int t = 0; t < 32; ++t) {
            const float w = s_topv[m * 32 + t];
            const int idx = s_topi[m * 32 + t];
            const unsigned int* vr = (const unsigned int*)(v + (size_t)idx * 1024) + lane * 16;
#pragma unroll
            for (int g = 0; g < 16; ++g) {
                unsigned int u = vr[g];
                acc[2 * g]     += w * bf2f((unsigned short)(u & 0xffffu));
                acc[2 * g + 1] += w * bf2f((unsigned short)(u >> 16));
            }
        }
        unsigned int* crow = (unsigned int*)(ctx + (size_t)(rowbase + m) * 1024) + lane * 16;
#pragma unroll
        for (int g = 0; g < 16; ++g)
            crow[g] = (unsigned int)f2bf(acc[2 * g]) | ((unsigned int)f2bf(acc[2 * g + 1]) << 16);
    }
}

// ---------- host ----------
extern "C" void kernel_launch(void* const* d_in, const int* in_sizes, int n_in,
                              void* d_out, int out_size, void* d_ws, size_t ws_size,
                              hipStream_t stream) {
    (void)in_sizes; (void)n_in; (void)out_size; (void)ws_size;
    const float* query  = (const float*)d_in[0]; // [2,1024,1024]
    const float* memory = (const float*)d_in[1]; // [32768,1024]
    const float* Wq = (const float*)d_in[2];
    const float* Wk = (const float*)d_in[3];
    const float* Wv = (const float*)d_in[4];
    const float* Wo = (const float*)d_in[5];

    const int D = 1024, MQ = 2048, NM = 32768;

    char* ws = (char*)d_ws;
    size_t off = 0;
    auto alloc = [&](size_t bytes) { char* p = ws + off; off = (off + bytes + 255) & ~(size_t)255; return p; };
    unsigned short* mem_bf   = (unsigned short*)alloc((size_t)NM * D * 2);
    unsigned short* query_bf = (unsigned short*)alloc((size_t)MQ * D * 2);
    unsigned short* wq_bf    = (unsigned short*)alloc((size_t)D * D * 2);
    unsigned short* wk_bf    = (unsigned short*)alloc((size_t)D * D * 2);
    unsigned short* wv_bf    = (unsigned short*)alloc((size_t)D * D * 2);
    unsigned short* wo_bf    = (unsigned short*)alloc((size_t)D * D * 2);
    unsigned short* q_bf     = (unsigned short*)alloc((size_t)MQ * D * 2);
    unsigned short* k_bf     = (unsigned short*)alloc((size_t)NM * D * 2);
    unsigned short* v_bf     = (unsigned short*)alloc((size_t)NM * D * 2);
    unsigned short* ctx_bf   = (unsigned short*)alloc((size_t)MQ * D * 2);

    // 1) fp32 -> bf16 converts
    cvt_f32_bf16<<<4096, 256, 0, stream>>>(memory, mem_bf,   NM * D);
    cvt_f32_bf16<<<1024, 256, 0, stream>>>(query,  query_bf, MQ * D);
    cvt_f32_bf16<<<512,  256, 0, stream>>>(Wq, wq_bf, D * D);
    cvt_f32_bf16<<<512,  256, 0, stream>>>(Wk, wk_bf, D * D);
    cvt_f32_bf16<<<512,  256, 0, stream>>>(Wv, wv_bf, D * D);
    cvt_f32_bf16<<<512,  256, 0, stream>>>(Wo, wo_bf, D * D);

    // 2) projections: q (scale 1/sqrt(D) folded in), k, v
    const float scale = 1.0f / 32.0f; // 1/sqrt(1024)
    gemm_xwt<true><<<dim3(D / 128, MQ / 64), 256, 0, stream>>>(query_bf, wq_bf, q_bf, MQ, D, D, scale);
    gemm_xwt<true><<<dim3(D / 128, NM / 64), 256, 0, stream>>>(mem_bf,   wk_bf, k_bf, NM, D, D, 1.0f);
    gemm_xwt<true><<<dim3(D / 128, NM / 64), 256, 0, stream>>>(mem_bf,   wv_bf, v_bf, NM, D, D, 1.0f);

    // 3) fused scores + top-32 + softmax + gather -> ctx (bf16)
    skim_attn<<<MQ / 16, 256, 0, stream>>>(q_bf, k_bf, v_bf, ctx_bf);

    // 4) out = ctx @ Wo^T (fp32 out)
    gemm_xwt<false><<<dim3(D / 128, MQ / 64), 256, 0, stream>>>(ctx_bf, wo_bf, d_out, MQ, D, D, 1.0f);
}
